// VectorQuantizer_910533066799
// MI455X (gfx1250) — matrix operand order repaired and checked by reading, not compiled, not run on hardware
//
#include <hip/hip_runtime.h>

#define K_CODES 8192
#define DIM 512
#define BN 32768              // 8 * 4096 rows
#define BETA 0.1f
#define TILE_N 32             // codes per LDS stage (two 16-code subtiles)
#define N_STAGES (K_CODES / TILE_N)   // 256
#define ROWS_PER_BLOCK 128
#define LDS_STRIDE 520        // 512 + 8 bf16 pad -> conflict-free row access
#define LDS_TILE (TILE_N * LDS_STRIDE)          // elements per buffer
#define LDS_BYTES (2 * LDS_TILE * 2)            // double buffered, bf16

typedef __attribute__((ext_vector_type(16))) __bf16 v16bf;
typedef __attribute__((ext_vector_type(8)))  float  v8f;

// Load 8 consecutive floats, convert to 8 bf16 (RNE via hardware cvt).
__device__ __forceinline__ void cvt8(const float* __restrict__ p, __bf16* d) {
    float4 a = ((const float4*)p)[0];
    float4 b = ((const float4*)p)[1];
    d[0] = (__bf16)a.x; d[1] = (__bf16)a.y; d[2] = (__bf16)a.z; d[3] = (__bf16)a.w;
    d[4] = (__bf16)b.x; d[5] = (__bf16)b.y; d[6] = (__bf16)b.z; d[7] = (__bf16)b.w;
}

// -------- Kernel 1: henorm[k] = 0.5 * ||embed_k||^2 ; zero loss accumulator ----
__global__ void vq_henorm_kernel(const float* __restrict__ embed,
                                 float* __restrict__ henorm,
                                 float* __restrict__ ws_loss) {
    const int wave = threadIdx.x >> 5;
    const int lane = threadIdx.x & 31;
    const int k = blockIdx.x * 8 + wave;
    const float* row = embed + (size_t)k * DIM;
    float s = 0.f;
#pragma unroll
    for (int i = 0; i < DIM / 32; ++i) {
        float v = row[lane + i * 32];
        s += v * v;
    }
#pragma unroll
    for (int m = 16; m >= 1; m >>= 1) s += __shfl_xor(s, m, 32);
    if (lane == 0) henorm[k] = 0.5f * s;
    if (blockIdx.x == 0 && threadIdx.x == 0) *ws_loss = 0.f;
}

// -------- Kernel 2: WMMA argmin over codebook ----------------------------------
// grid = BN/128 blocks, 256 threads (8 wave32). Wave w owns rows
// [blk*128 + w*16, +16). Codes streamed 32 at a time via double-buffered LDS;
// each wave runs TWO independent WMMA accumulator chains (codes c, c+16).
__global__ void __launch_bounds__(256)
vq_argmin_kernel(const float* __restrict__ z,
                 const float* __restrict__ embed,
                 const float* __restrict__ henorm,
                 int*   __restrict__ codes_i,
                 float* __restrict__ out_codes) {
    extern __shared__ __align__(16) __bf16 Btile[];   // 2 * 32 * LDS_STRIDE bf16

    const int tid    = threadIdx.x;
    const int wave   = tid >> 5;
    const int lane   = tid & 31;
    const int lane15 = lane & 15;
    const int hi     = lane >> 4;         // 0: lanes 0-15, 1: lanes 16-31
    const int m0     = blockIdx.x * ROWS_PER_BLOCK + wave * 16;

    // ---- A fragments: this wave's 16 rows x 512, bf16, persistent in VGPRs ----
    // 16-bit A 16x32 layout: lane<16 holds K {k0..k0+7, k0+16..k0+23} of row lane;
    // lane>=16 holds the +8 shifted ranges (ISA 7.12.2).
    v16bf afrag[16];
    {
        const float* zp = z + (size_t)(m0 + lane15) * DIM + hi * 8;
#pragma unroll
        for (int f = 0; f < 16; ++f) {
            union { v16bf v; __bf16 e[16]; } u;
            cvt8(zp + f * 32,      u.e);
            cvt8(zp + f * 32 + 16, u.e + 8);
            afrag[f] = u.v;
        }
    }

    // ---- B tile staging: 32 codes x 512 f32 -> bf16 in LDS (padded rows) ------
    // thread handles chunks c=0..7: tile row = c*4 + tid/64, cols [ (tid%64)*8, +8 )
    const int rowbase = tid >> 6;          // 0..3
    const int col     = (tid & 63) * 8;    // 0..504, 16B aligned

    uint4 pre[8];
    auto fetch = [&](int t) {
        const float* base = embed + (size_t)t * TILE_N * DIM;
#pragma unroll
        for (int c = 0; c < 8; ++c) {
            const int row = c * 4 + rowbase;
            union { __bf16 h[8]; uint4 q; } w;
            cvt8(base + row * DIM + col, w.h);
            pre[c] = w.q;
        }
    };
    auto stage = [&](int buf) {
        __bf16* bb = Btile + buf * LDS_TILE;
#pragma unroll
        for (int c = 0; c < 8; ++c) {
            const int row = c * 4 + rowbase;
            *(uint4*)(bb + row * LDS_STRIDE + col) = pre[c];
        }
    };

    float best[8];
    int   bidx[8];
#pragma unroll
    for (int j = 0; j < 8; ++j) { best[j] = -3.0e38f; bidx[j] = 0; }

    fetch(0);
    stage(0);
    __syncthreads();

    for (int t = 0; t < N_STAGES; ++t) {
        const int buf = t & 1;
        if (t + 1 < N_STAGES) fetch(t + 1);   // overlap next stage's global loads
        if (t + 2 < N_STAGES)                  // warm L2/WGP$ two stages ahead
            __builtin_prefetch(embed + (size_t)(t + 2) * TILE_N * DIM
                               + (size_t)rowbase * DIM + col, 0, 1);

        v8f acc0 = {};
        v8f acc1 = {};
        // 16-bit B 32x16 layout: lane<16 holds K {0..7, 16..23} of column lane,
        // lane>=16 holds K {8..15, 24..31} (same order as A).
        const __bf16* bp0 = Btile + buf * LDS_TILE + lane15 * LDS_STRIDE + hi * 8;
        const __bf16* bp1 = bp0 + 16 * LDS_STRIDE;
#pragma unroll
        for (int f = 0; f < 16; ++f) {
            v16bf b0, b1;
            ((uint4*)&b0)[0] = *(const uint4*)(bp0 + f * 32);
            ((uint4*)&b0)[1] = *(const uint4*)(bp0 + f * 32 + 16);
            ((uint4*)&b1)[0] = *(const uint4*)(bp1 + f * 32);
            ((uint4*)&b1)[1] = *(const uint4*)(bp1 + f * 32 + 16);
            acc0 = __builtin_amdgcn_wmma_f32_16x16x32_bf16(
                false, afrag[f], false, b0, (short)0, acc0, false, false);
            acc1 = __builtin_amdgcn_wmma_f32_16x16x32_bf16(
                false, afrag[f], false, b1, (short)0, acc1, false, false);
        }

        // score = dot - 0.5*||e||^2 ; argmin dist == argmax score
        const float he0 = henorm[t * TILE_N + lane15];
        const float he1 = henorm[t * TILE_N + 16 + lane15];
        const int  c0   = t * TILE_N + lane15;
        const int  c1   = c0 + 16;
#pragma unroll
        for (int j = 0; j < 8; ++j) {
            float v0 = acc0[j] - he0;
            if (v0 > best[j]) { best[j] = v0; bidx[j] = c0; }   // strict > keeps first
            float v1 = acc1[j] - he1;
            if (v1 > best[j]) { best[j] = v1; bidx[j] = c1; }
        }

        __syncthreads();
        if (t + 1 < N_STAGES) stage(buf ^ 1);
        __syncthreads();
    }

    // ---- cross-lane argmax within each 16-lane row group ----------------------
#pragma unroll
    for (int j = 0; j < 8; ++j) {
#pragma unroll
        for (int s = 1; s < 16; s <<= 1) {
            float ov = __shfl_xor(best[j], s, 32);
            int   oi = __shfl_xor(bidx[j], s, 32);
            if (ov > best[j] || (ov == best[j] && oi < bidx[j])) {
                best[j] = ov; bidx[j] = oi;
            }
        }
    }
    // C layout: VGPR j = row j (lanes 0-15) / row 8+j (lanes 16-31)
    if (lane15 == 0) {
        const int base = m0 + hi * 8;
#pragma unroll
        for (int j = 0; j < 8; ++j) {
            codes_i[base + j]   = bidx[j];
            out_codes[base + j] = (float)bidx[j];
        }
    }
}

// -------- Kernel 3: gather z_q rows + commit-loss partial sums -----------------
__global__ void vq_gather_loss_kernel(const float* __restrict__ z,
                                      const float* __restrict__ embed,
                                      const int*   __restrict__ codes_i,
                                      float* __restrict__ out_zq,
                                      float* __restrict__ ws_loss) {
    __shared__ float ls[4];
    const int row = blockIdx.x;
    const int t   = threadIdx.x;       // 128 threads, 4 floats each
    const int code = codes_i[row];

    const float4* e4 = (const float4*)(embed + (size_t)code * DIM);
    const float4* z4 = (const float4*)(z + (size_t)row * DIM);
    float4*       o4 = (float4*)(out_zq + (size_t)row * DIM);

    float4 ev = e4[t];
    float4 zv = z4[t];
    o4[t] = ev;                        // z_q_st == z_q numerically

    float dx = zv.x - ev.x, dy = zv.y - ev.y, dz = zv.z - ev.z, dw = zv.w - ev.w;
    float s = dx * dx + dy * dy + dz * dz + dw * dw;
#pragma unroll
    for (int m = 16; m >= 1; m >>= 1) s += __shfl_xor(s, m, 32);
    if ((t & 31) == 0) ls[t >> 5] = s;
    __syncthreads();
    if (t == 0) atomicAdd(ws_loss, ls[0] + ls[1] + ls[2] + ls[3]);
}

// -------- Kernel 4: finalize loss ----------------------------------------------
__global__ void vq_finalize_kernel(const float* __restrict__ ws_loss,
                                   float* __restrict__ out_loss) {
    if (threadIdx.x == 0 && blockIdx.x == 0)
        *out_loss = BETA * (*ws_loss) * (1.0f / 16777216.0f);  // / (BN*DIM)
}

extern "C" void kernel_launch(void* const* d_in, const int* in_sizes, int n_in,
                              void* d_out, int out_size, void* d_ws, size_t ws_size,
                              hipStream_t stream) {
    const float* z     = (const float*)d_in[0];   // (8,4096,512) f32
    const float* embed = (const float*)d_in[1];   // (8192,512)   f32

    float* out       = (float*)d_out;
    float* out_zq    = out;                               // BN*DIM
    float* out_codes = out + (size_t)BN * DIM;            // BN (codes as float)
    float* out_loss  = out_codes + BN;                    // scalar

    // workspace: henorm (K f32) | codes_i (BN i32) | loss accumulator (f32)
    float* henorm  = (float*)d_ws;
    int*   codes_i = (int*)((char*)d_ws + (size_t)K_CODES * sizeof(float));
    float* ws_loss = (float*)((char*)d_ws + (size_t)K_CODES * sizeof(float)
                              + (size_t)BN * sizeof(int));

    vq_henorm_kernel<<<K_CODES / 8, 256, 0, stream>>>(embed, henorm, ws_loss);
    vq_argmin_kernel<<<BN / ROWS_PER_BLOCK, 256, LDS_BYTES, stream>>>(
        z, embed, henorm, codes_i, out_codes);
    vq_gather_loss_kernel<<<BN, 128, 0, stream>>>(z, embed, codes_i, out_zq, ws_loss);
    vq_finalize_kernel<<<1, 1, 0, stream>>>(ws_loss, out_loss);
}
